// AdapterController_6408091205681
// MI455X (gfx1250) — matrix operand order repaired and checked by reading, not compiled, not run on hardware
//
#include <hip/hip_runtime.h>

// MI455X (gfx1250) fused routed-adapter kernel.
// bf16 WMMA (f32 accumulate) for both GEMMs, pre-transposed bf16 weights in
// workspace, async global->LDS weight staging (ASYNCcnt path) when available,
// fused bias/relu/residual/LayerNorm. Memory-bound target: ~270 MB @ 23.3 TB/s.

typedef __attribute__((ext_vector_type(16))) __bf16 v16bf;
typedef __attribute__((ext_vector_type(8)))  float  v8f;
typedef __attribute__((ext_vector_type(4)))  __bf16 bf16x4;
typedef __attribute__((ext_vector_type(4)))  int    v4i_t;

#if __has_builtin(__builtin_amdgcn_global_load_async_to_lds_b128)
#define USE_ASYNC_STAGE 1
#else
#define USE_ASYNC_STAGE 0
#endif

namespace {
constexpr int B = 16, S = 2048, H = 1024, K = 128, P = 8;
constexpr int TM   = 32;    // token rows per workgroup
constexpr int WSTR = 136;   // padded LDS stride for transposed weight chunk
constexpr float EPS = 1e-5f;

constexpr int XT_ELEMS = TM * H;      // bf16 staged activations
constexpr int ZT_ELEMS = TM * K;      // bf16 bottleneck
constexpr int WT_ELEMS = 128 * WSTR;  // bf16 weight chunk, transposed
constexpr int YT_ELEMS = TM * H;      // f32 pre-norm output
constexpr size_t SMEM_BYTES =
    (size_t)(XT_ELEMS + ZT_ELEMS + WT_ELEMS) * 2 +
    (size_t)(YT_ELEMS + 256 + 256 + 64) * 4;   // ~236 KB < 320 KB/WGP
}

// ---- address-space helpers for the async-to-LDS builtin ----
// The builtin takes b128-typed (int4) pointers: AS1 source, AS3 LDS dest.
// generic->local addrspacecast on amdgcn is a low-32-bit truncation, and
// generic->global keeps the 64-bit address, so the integer route is exact.
#if USE_ASYNC_STAGE
typedef __attribute__((address_space(1))) v4i_t as1_v4i;
typedef __attribute__((address_space(3))) v4i_t as3_v4i;
__device__ __forceinline__ as1_v4i* to_as1(const void* p) {
  return (as1_v4i*)(unsigned long long)p;
}
__device__ __forceinline__ as3_v4i* to_as3(void* p) {
  return (as3_v4i*)(unsigned int)(unsigned long long)p;
}
#endif

__device__ __forceinline__ void wait_async_lds() {
#if USE_ASYNC_STAGE
#if __has_builtin(__builtin_amdgcn_s_wait_asynccnt)
  __builtin_amdgcn_s_wait_asynccnt(0);
#else
  asm volatile("s_wait_asynccnt 0x0" ::: "memory");
#endif
#endif
}

// Stage a 128x128 bf16 weight chunk (rows contiguous along reduction dim in
// the workspace) into LDS with padded row stride WSTR. 2048 x 16B transfers.
__device__ __forceinline__ void stage_weight_chunk(const __bf16* __restrict__ src,
                                                   int srcStride, __bf16* wT,
                                                   int tid) {
#pragma unroll
  for (int it = 0; it < 8; ++it) {
    const int i   = tid + it * 256;
    const int col = i >> 4;            // 0..127
    const int seg = (i & 15) * 8;      // bf16 elems, 16B granules
    const __bf16* s = src + (size_t)col * srcStride + seg;
    __bf16*       d = wT + col * WSTR + seg;
#if USE_ASYNC_STAGE
    __builtin_amdgcn_global_load_async_to_lds_b128(to_as1(s), to_as3(d), 0, 0);
#else
    *(uint4*)d = *(const uint4*)s;
#endif
  }
}

// A fragment (16x32 bf16, wave32): lane L holds row (L&15); K group base is
// 0 for lanes 0-15, 8 for lanes 16-31; VGPR v packs K pair starting at
// 2v (+8 extra for v>=4). Each u32 read grabs a contiguous bf16 pair.
__device__ __forceinline__ v16bf load_frag_a(const __bf16* base, int rowBase,
                                             int rowStride, int kOff, int lane) {
  union { unsigned int u[8]; v16bf v; } r;
  const int row = rowBase + (lane & 15);
  const int kg  = (lane < 16) ? 0 : 8;
  const __bf16* rp = base + row * rowStride;
#pragma unroll
  for (int v = 0; v < 8; ++v) {
    const int k = kOff + kg + 2 * v + ((v >= 4) ? 8 : 0);
    r.u[v] = *(const unsigned int*)(rp + k);
  }
  return r.v;
}

// B fragment (32x16 bf16, wave32): lane L holds col (L&15); K group base is
// 0 for lanes 0-15, 8 for lanes 16-31; VGPR v packs K pair starting at
// 2v (+8 extra for v>=4).
// Weight chunk is stored col-major in LDS so the pair is contiguous.
__device__ __forceinline__ v16bf load_frag_b(const __bf16* base, int colBase,
                                             int colStride, int kOff, int lane) {
  union { unsigned int u[8]; v16bf v; } r;
  const int col = colBase + (lane & 15);
  const int kg  = (lane < 16) ? 0 : 8;
  const __bf16* cp = base + col * colStride;
#pragma unroll
  for (int v = 0; v < 8; ++v) {
    r.u[v] = *(const unsigned int*)(cp + kOff + kg + 2 * v + ((v >= 4) ? 8 : 0));
  }
  return r.v;
}

// ---- prologue: f32 -> bf16 AND transpose both weight tensors ----
//   dT[p][kcol][h] = down_w[p][h][kcol]   (contiguous along h)
//   uT[p][ncol][k] = up_w[p][k][ncol]     (contiguous along k)
__global__ void wprep_kernel(const float* __restrict__ dw,
                             const float* __restrict__ uw,
                             __bf16* __restrict__ dT,
                             __bf16* __restrict__ uT, int n4) {
  int i = blockIdx.x * blockDim.x + threadIdx.x;
  if (i >= n4) return;
  const int idx = i * 4;
  {
    const int h0  = idx & (H - 1);
    const int col = (idx >> 10) & (K - 1);
    const int p   = idx >> 17;
    bf16x4 o;
#pragma unroll
    for (int j = 0; j < 4; ++j)
      o[j] = (__bf16)dw[(size_t)((p * H + h0 + j) << 7) + col];
    *(bf16x4*)(dT + idx) = o;
  }
  {
    const int k0 = idx & (K - 1);
    const int n  = (idx >> 7) & (H - 1);
    const int p  = idx >> 17;
    bf16x4 o;
#pragma unroll
    for (int j = 0; j < 4; ++j)
      o[j] = (__bf16)uw[(size_t)((p * K + k0 + j) << 10) + n];
    *(bf16x4*)(uT + idx) = o;
  }
}

__global__ __launch_bounds__(256)
void adapter_kernel(const float* __restrict__ hidden,
                    const int* __restrict__ pids,
                    const __bf16* __restrict__ wbf,   // dT [P*K*H] ++ uT [P*H*K]
                    const float* __restrict__ down_b,
                    const float* __restrict__ up_b,
                    const float* __restrict__ ln_g,
                    const float* __restrict__ ln_b,
                    float* __restrict__ out) {
  extern __shared__ char smem[];
  __bf16* xT = (__bf16*)smem;            // [TM][H]   bf16 activations
  __bf16* zT = xT + XT_ELEMS;            // [TM][K]   bf16 bottleneck
  __bf16* wT = zT + ZT_ELEMS;            // [128][WSTR] bf16 weight chunk (T)
  float*  yT   = (float*)(wT + WT_ELEMS);// [TM][H]   f32 pre-norm
  float*  red  = yT + YT_ELEMS;          // [256] row partial sums
  float*  red2 = red + 256;              // [256] row partial sumsq
  float*  stats = red2 + 256;            // [TM][2] mean, rstd

  const int tid  = threadIdx.x;
  const int lane = tid & 31;
  const int wv   = tid >> 5;             // wave id 0..7
  const int p    = pids[blockIdx.y];
  const size_t rowBase = (size_t)blockIdx.y * S + (size_t)blockIdx.x * TM;

  const __bf16* dTp = wbf + (size_t)p * K * H;                      // [K][H]
  const __bf16* uTp = wbf + (size_t)P * K * H + (size_t)p * H * K;  // [H][K]

  // ---- stage activation tile, f32 -> bf16 ----
  {
    const float4* xg = (const float4*)(hidden + rowBase * H);
#pragma unroll
    for (int it = 0; it < (TM * H / 4) / 256; ++it) {
      const int i = tid + it * 256;
      float4 v = xg[i];
      bf16x4 o;
      o[0] = (__bf16)v.x; o[1] = (__bf16)v.y; o[2] = (__bf16)v.z; o[3] = (__bf16)v.w;
      *(bf16x4*)(xT + i * 4) = o;
    }
  }

  // ---- down-proj: Z = relu(X @ Wd + bd) via v_wmma_f32_16x16x32_bf16 ----
  v8f accZ0 = {}, accZ1 = {};
  for (int hc = 0; hc < H / 128; ++hc) {
    __syncthreads();                      // prior wT readers done (and xT staged)
    stage_weight_chunk(dTp + hc * 128, H, wT, tid);
    wait_async_lds();
    __syncthreads();
#pragma unroll
    for (int kk = 0; kk < 4; ++kk) {
      v16bf bfr = load_frag_b(wT, wv * 16, WSTR, kk * 32, lane);
      v16bf a0  = load_frag_a(xT, 0,  H, hc * 128 + kk * 32, lane);
      v16bf a1  = load_frag_a(xT, 16, H, hc * 128 + kk * 32, lane);
      accZ0 = __builtin_amdgcn_wmma_f32_16x16x32_bf16(false, a0, false, bfr,
                                                      (short)0, accZ0, false, false);
      accZ1 = __builtin_amdgcn_wmma_f32_16x16x32_bf16(false, a1, false, bfr,
                                                      (short)0, accZ1, false, false);
    }
  }
  // bias + relu, pack bottleneck to LDS (C/D layout: M = r + 8*(lane>=16))
  {
    const int col  = wv * 16 + (lane & 15);
    const float bd = down_b[p * K + col];
    const int rofs = (lane >= 16) ? 8 : 0;
#pragma unroll
    for (int r = 0; r < 8; ++r) {
      float z0 = accZ0[r] + bd; z0 = z0 > 0.f ? z0 : 0.f;
      float z1 = accZ1[r] + bd; z1 = z1 > 0.f ? z1 : 0.f;
      zT[(r + rofs) * K + col]      = (__bf16)z0;
      zT[(16 + r + rofs) * K + col] = (__bf16)z1;
    }
  }

  // ---- up-proj + residual + bias, chunked over output columns ----
  for (int nc = 0; nc < H / 128; ++nc) {
    __syncthreads();                      // prior wT readers done / zT visible
    stage_weight_chunk(uTp + (size_t)nc * 128 * K, K, wT, tid);
    wait_async_lds();
    __syncthreads();
    v8f accY0 = {}, accY1 = {};
#pragma unroll
    for (int kk = 0; kk < 4; ++kk) {
      v16bf bfr = load_frag_b(wT, wv * 16, WSTR, kk * 32, lane);
      v16bf a0  = load_frag_a(zT, 0,  K, kk * 32, lane);
      v16bf a1  = load_frag_a(zT, 16, K, kk * 32, lane);
      accY0 = __builtin_amdgcn_wmma_f32_16x16x32_bf16(false, a0, false, bfr,
                                                      (short)0, accY0, false, false);
      accY1 = __builtin_amdgcn_wmma_f32_16x16x32_bf16(false, a1, false, bfr,
                                                      (short)0, accY1, false, false);
    }
    {
      const int colC = nc * 128 + wv * 16 + (lane & 15);
      const float bu = up_b[p * H + colC];
      const int rofs = (lane >= 16) ? 8 : 0;
#pragma unroll
      for (int r = 0; r < 8; ++r) {
        const int row0 = r + rofs, row1 = 16 + r + rofs;
        // residual re-read from f32 hidden (L2-resident) for full precision
        float y0 = accY0[r] + bu + hidden[(rowBase + row0) * H + colC];
        float y1 = accY1[r] + bu + hidden[(rowBase + row1) * H + colC];
        yT[row0 * H + colC] = y0;
        yT[row1 * H + colC] = y1;
      }
    }
  }
  __syncthreads();

  // ---- fused LayerNorm: 8 threads per row ----
  {
    const int row = tid >> 3, sub = tid & 7;
    const float4* yv = (const float4*)(yT + row * H + sub * 128);
    float s = 0.f, ss = 0.f;
#pragma unroll 8
    for (int j = 0; j < 32; ++j) {
      float4 t = yv[j];
      s  += t.x + t.y + t.z + t.w;
      ss += t.x * t.x + t.y * t.y + t.z * t.z + t.w * t.w;
    }
    red[tid] = s; red2[tid] = ss;
    __syncthreads();
    if (sub == 0) {
      float s1 = 0.f, s2 = 0.f;
#pragma unroll
      for (int i = 0; i < 8; ++i) { s1 += red[row * 8 + i]; s2 += red2[row * 8 + i]; }
      const float mu  = s1 * (1.f / H);
      const float var = s2 * (1.f / H) - mu * mu;
      stats[row * 2 + 0] = mu;
      stats[row * 2 + 1] = rsqrtf(var + EPS);
    }
    __syncthreads();
    const float mu = stats[row * 2 + 0], rs = stats[row * 2 + 1];
    const float* gp = ln_g + p * H + sub * 128;
    const float* bp = ln_b + p * H + sub * 128;
    float4* op = (float4*)(out + (rowBase + row) * H + sub * 128);
#pragma unroll 8
    for (int j = 0; j < 32; ++j) {
      float4 t = yv[j];
      const int c = j * 4;
      t.x = (t.x - mu) * rs * gp[c + 0] + bp[c + 0];
      t.y = (t.y - mu) * rs * gp[c + 1] + bp[c + 1];
      t.z = (t.z - mu) * rs * gp[c + 2] + bp[c + 2];
      t.w = (t.w - mu) * rs * gp[c + 3] + bp[c + 3];
      op[j] = t;
    }
  }
}

extern "C" void kernel_launch(void* const* d_in, const int* in_sizes, int n_in,
                              void* d_out, int out_size, void* d_ws, size_t ws_size,
                              hipStream_t stream) {
  const float* hidden = (const float*)d_in[0];
  const int*   pids   = (const int*)d_in[1];
  const float* down_w = (const float*)d_in[2];
  const float* down_b = (const float*)d_in[3];
  const float* up_w   = (const float*)d_in[4];
  const float* up_b   = (const float*)d_in[5];
  const float* ln_g   = (const float*)d_in[6];
  const float* ln_b   = (const float*)d_in[7];
  float* out = (float*)d_out;
  __bf16* wbf = (__bf16*)d_ws;          // 4 MiB: dT (transposed down) ++ uT

  const int nEach = P * H * K;          // 1,048,576 per tensor
  const int n4 = nEach / 4;
  wprep_kernel<<<(n4 + 255) / 256, 256, 0, stream>>>(down_w, up_w, wbf,
                                                     wbf + nEach, n4);

  (void)hipFuncSetAttribute((const void*)adapter_kernel,
                            hipFuncAttributeMaxDynamicSharedMemorySize,
                            (int)SMEM_BYTES);
  dim3 grid(S / TM, B);
  adapter_kernel<<<grid, 256, SMEM_BYTES, stream>>>(hidden, pids, wbf, down_b,
                                                    up_b, ln_g, ln_b, out);
  (void)in_sizes; (void)n_in; (void)out_size; (void)ws_size;
}
